// ProtoSAE_22539988369827
// MI455X (gfx1250) — matrix operand order repaired and checked by reading, not compiled, not run on hardware
//
#include <hip/hip_runtime.h>
#include <hip/hip_bf16.h>

// CDNA5 / gfx1250 fused ProtoSAE forward.
// h = relu(BN(x @ W1[l]^T + b1)); enc = h @ W2[l]^T + b2; act = logmap(||enc-proto||^2)
// bf16 WMMA (16x16x32, f32 accum), double-buffered LDS staging (1 barrier per
// K-step), fully fused so the 536MB f32 intermediate never leaves the WGP.

typedef __attribute__((ext_vector_type(16))) __bf16 v16bf;
typedef __attribute__((ext_vector_type(2)))  __bf16 v2bf;
typedef __attribute__((ext_vector_type(8)))  float  v8f;
typedef __attribute__((ext_vector_type(2)))  float  v2f;

#define NI_DIM   1024
#define LH_DIM   128    // hidden per latent
#define PP_DIM   64     // prototype dim
#define TOK_TILE 64
#define KB       32     // K step for GEMM1
#define NSTEPS   (NI_DIM / KB)
#define XS_STR   40     // 32 + 8 pad (bf16 elems) -> 80B row stride, 16B aligned
#define WS_STR   40
#define HS_STR   136    // 128 + 8 pad -> 272B row stride, 16B aligned

#define XS_ELE   (TOK_TILE * XS_STR)     // 2560 elems per buffer
#define WS_ELE   (LH_DIM   * WS_STR)     // 5120 elems per buffer
#define R0_ELE   (2*XS_ELE + 2*WS_ELE)   // staging region (aliased by hs later)
#define HS_ELE   (TOK_TILE * HS_STR)     // 8704 (fits inside R0_ELE = 15360)
#define W2_ELE   (PP_DIM   * HS_STR)     // 8704

// Packed f32x2 -> bf16x2: lowers to one v_cvt_pk_bf16_f32.
__device__ __forceinline__ unsigned int f2bf2(float lo, float hi) {
    v2f f; f.x = lo; f.y = hi;
    union { v2bf b; unsigned int u; } c;
    c.b = __builtin_convertvector(f, v2bf);
    return c.u;
}
__device__ __forceinline__ unsigned short f2bf(float f) {
    union { __bf16 b; unsigned short u; } c;
    c.b = (__bf16)f;
    return c.u;
}
__device__ __forceinline__ void cvt4(unsigned int& d0, unsigned int& d1, float4 a) {
    d0 = f2bf2(a.x, a.y);
    d1 = f2bf2(a.z, a.w);
}

union Frag { v16bf v; uint4 q[2]; };

__device__ __forceinline__ void ld_frag(Frag& f, const unsigned short* p, int second_off) {
    f.q[0] = *(const uint4*)(p);
    f.q[1] = *(const uint4*)(p + second_off);
}

__global__ __launch_bounds__(256)
void proto_sae_fused(const float* __restrict__ x,
                     const float* __restrict__ W1,
                     const float* __restrict__ b1,
                     const float* __restrict__ gamma,
                     const float* __restrict__ beta,
                     const float* __restrict__ rmean,
                     const float* __restrict__ rvar,
                     const float* __restrict__ W2,
                     const float* __restrict__ b2,
                     const float* __restrict__ protos,
                     float* __restrict__ out)
{
    __shared__ __align__(16) unsigned short smem[R0_ELE + W2_ELE];
    __shared__ float scaleS[LH_DIM], shiftS[LH_DIM], pbS[PP_DIM], sums[TOK_TILE];

    unsigned short* const xs  = smem;                // ping-pong x tiles
    unsigned short* const ws  = smem + 2*XS_ELE;     // ping-pong W1 tiles
    unsigned short* const hs  = smem;                // aliases staging (post-GEMM1)
    unsigned short* const w2s = smem + R0_ELE;       // W2[l] bf16

    const int t    = threadIdx.x;
    const int lane = t & 31;
    const int w    = t >> 5;
    const int mw   = w >> 1;          // M-subtile (0..3)
    const int nb   = (w & 1) * 4;     // GEMM1 N-subtile base (0 or 4)
    const int T0   = blockIdx.x * TOK_TILE;
    const int l    = blockIdx.y;

    const int am = lane & 15;         // A: row-in-tile / B: col-in-tile
    const int kh = lane >> 4;         // K half selector

    // ---- per-latent parameter prep: fold b1 + BN into scale/shift ----
    if (t < LH_DIM) {
        float g  = gamma[l*LH_DIM + t];
        float be = beta [l*LH_DIM + t];
        float mn = rmean[l*LH_DIM + t];
        float vr = rvar [l*LH_DIM + t];
        float bb = b1   [l*LH_DIM + t];
        float sc = g * rsqrtf(vr + 1e-5f);
        scaleS[t] = sc;
        shiftS[t] = (bb - mn) * sc + be;
    }
    if (t < PP_DIM) pbS[t] = b2[l*PP_DIM + t] - protos[l*PP_DIM + t];

    // ---- W2[l] (64x128 f32) -> bf16 LDS, K contiguous per row ----
    {
        const int row = t >> 2;             // 0..63
        const int c   = (t & 3) * 32;       // 0,32,64,96
        const float* src = W2 + ((size_t)l * PP_DIM + row) * LH_DIM + c;
        #pragma unroll
        for (int i = 0; i < 4; ++i) {
            float4 a = *(const float4*)(src + i*8);
            float4 b = *(const float4*)(src + i*8 + 4);
            uint4 p;
            cvt4(p.x, p.y, a);
            cvt4(p.z, p.w, b);
            *(uint4*)(w2s + row*HS_STR + c + i*8) = p;
        }
    }

    // ---- GEMM1: [64 x 1024] @ [1024 x 128], double-buffered K pipeline ----
    const int xrow = t >> 2, xc = (t & 3) * 8;    // 4 threads/row, 8 f32 each
    const int wrow = t >> 1, wc = (t & 1) * 16;   // 2 threads/row, 16 f32 each
    const float* xg_base = x  + (size_t)(T0 + xrow) * NI_DIM + xc;
    const float* wg_base = W1 + ((size_t)l * LH_DIM + wrow) * NI_DIM + wc;

    float4 xg0, xg1, wg0, wg1, wg2, wg3;          // staged next slab (f32)

    auto load_slab = [&](int ks) {
        const float* xb = xg_base + ks * KB;
        const float* wb = wg_base + ks * KB;
        xg0 = *(const float4*)(xb);     xg1 = *(const float4*)(xb + 4);
        wg0 = *(const float4*)(wb);     wg1 = *(const float4*)(wb + 4);
        wg2 = *(const float4*)(wb + 8); wg3 = *(const float4*)(wb + 12);
    };
    auto store_slab = [&](int buf) {              // cvt_pk + b128 stores
        unsigned short* xd = xs + buf*XS_ELE + xrow*XS_STR + xc;
        unsigned short* wd = ws + buf*WS_ELE + wrow*WS_STR + wc;
        uint4 p;
        cvt4(p.x, p.y, xg0); cvt4(p.z, p.w, xg1);
        *(uint4*)xd = p;
        uint4 q0, q1;
        cvt4(q0.x, q0.y, wg0); cvt4(q0.z, q0.w, wg1);
        cvt4(q1.x, q1.y, wg2); cvt4(q1.z, q1.w, wg3);
        *(uint4*)(wd)     = q0;
        *(uint4*)(wd + 8) = q1;
    };

    const v8f zero8 = {0.f,0.f,0.f,0.f,0.f,0.f,0.f,0.f};
    v8f acc0 = zero8, acc1 = zero8, acc2 = zero8, acc3 = zero8;

    const int aoff = (mw*16 + am) * XS_STR + kh*16;
    const int boff = (nb*16 + am) * WS_STR + kh*16;

    load_slab(0);
    store_slab(0);
    load_slab(1);
    __syncthreads();

    for (int ks = 0; ks < NSTEPS; ++ks) {
        const unsigned short* xbuf = xs + (ks & 1) * XS_ELE;
        const unsigned short* wbuf = ws + (ks & 1) * WS_ELE;

        // issue ALL fragment loads first, then the 4 WMMAs (partial waits)
        Frag a, b0, b1f, b2f, b3;
        ld_frag(a,   xbuf + aoff,                 8);
        ld_frag(b0,  wbuf + boff + 0*16*WS_STR,   8);
        ld_frag(b1f, wbuf + boff + 1*16*WS_STR,   8);
        ld_frag(b2f, wbuf + boff + 2*16*WS_STR,   8);
        ld_frag(b3,  wbuf + boff + 3*16*WS_STR,   8);

        acc0 = __builtin_amdgcn_wmma_f32_16x16x32_bf16(false, a.v, false, b0.v,  (short)0, acc0, false, false);
        acc1 = __builtin_amdgcn_wmma_f32_16x16x32_bf16(false, a.v, false, b1f.v, (short)0, acc1, false, false);
        acc2 = __builtin_amdgcn_wmma_f32_16x16x32_bf16(false, a.v, false, b2f.v, (short)0, acc2, false, false);
        acc3 = __builtin_amdgcn_wmma_f32_16x16x32_bf16(false, a.v, false, b3.v,  (short)0, acc3, false, false);

        // write slab ks+1 into the other buffer (overlaps the WMMAs above),
        // then start the global loads for slab ks+2.
        if (ks + 1 < NSTEPS) {
            store_slab((ks + 1) & 1);
            if (ks + 2 < NSTEPS) load_slab(ks + 2);
        }
        __syncthreads();
    }

    // ---- epilogue 1: BN + ReLU, bf16 h -> LDS (aliases dead staging) ----
    {
        const int rowoff = mw*16 + (kh << 3);
        #pragma unroll
        for (int j = 0; j < 4; ++j) {
            const int col = (nb + j) * 16 + am;
            const float sc = scaleS[col], sh = shiftS[col];
            v8f av = (j == 0) ? acc0 : (j == 1) ? acc1 : (j == 2) ? acc2 : acc3;
            #pragma unroll
            for (int r = 0; r < 8; ++r) {
                float v = av[r] * sc + sh;
                v = fmaxf(v, 0.f);
                hs[(rowoff + r) * HS_STR + col] = f2bf(v);
            }
        }
    }
    if (t < TOK_TILE) sums[t] = 0.f;
    __syncthreads();

    // ---- GEMM2: [64 x 128] @ [128 x 64], K in 4 steps ----
    v8f e0 = zero8, e1 = zero8;
    const int n2b = (w & 1) * 2;
    const unsigned short* apA2 = hs  + (mw*16  + am) * HS_STR + kh*16;
    const unsigned short* bpB2 = w2s + (n2b*16 + am) * HS_STR + kh*16;
    #pragma unroll
    for (int ks2 = 0; ks2 < 4; ++ks2) {
        Frag a2, c0, c1;
        ld_frag(a2, apA2 + ks2*32,                 8);
        ld_frag(c0, bpB2 + ks2*32 + 0*16*HS_STR,   8);
        ld_frag(c1, bpB2 + ks2*32 + 1*16*HS_STR,   8);
        e0 = __builtin_amdgcn_wmma_f32_16x16x32_bf16(false, a2.v, false, c0.v, (short)0, e0, false, false);
        e1 = __builtin_amdgcn_wmma_f32_16x16x32_bf16(false, a2.v, false, c1.v, (short)0, e1, false, false);
    }

    // ---- prototype L2 distance: cross-lane reduce over N, LDS accumulate ----
    #pragma unroll
    for (int j = 0; j < 2; ++j) {
        const int col = (n2b + j) * 16 + am;
        const float pbv = pbS[col];
        v8f ev = j ? e1 : e0;
        #pragma unroll
        for (int r = 0; r < 8; ++r) {
            float d = ev[r] + pbv;           // enc + b2 - proto
            float s = d * d;
            s += __shfl_xor(s, 1);
            s += __shfl_xor(s, 2);
            s += __shfl_xor(s, 4);
            s += __shfl_xor(s, 8);           // sum over 16-lane N group
            if (am == 0)
                atomicAdd(&sums[mw*16 + (kh << 3) + r], s);
        }
    }
    __syncthreads();

    // ---- log activation + store ----
    if (t < TOK_TILE) {
        float l2  = sums[t];
        float raw = __logf((l2 + 1.0f) / (l2 + 1e-4f));
        float act = raw * (300.0f / 9.210340371976184f) - 100.0f;  // /log(1e4)
        out[(size_t)(T0 + t) * 128 + l] = act;
    }
}

extern "C" void kernel_launch(void* const* d_in, const int* in_sizes, int n_in,
                              void* d_out, int out_size, void* d_ws, size_t ws_size,
                              hipStream_t stream) {
    (void)in_sizes; (void)n_in; (void)d_ws; (void)ws_size; (void)out_size;
    const float* x      = (const float*)d_in[0];
    const float* W1     = (const float*)d_in[1];
    const float* b1     = (const float*)d_in[2];
    const float* gamma  = (const float*)d_in[3];
    const float* beta   = (const float*)d_in[4];
    const float* rmean  = (const float*)d_in[5];
    const float* rvar   = (const float*)d_in[6];
    const float* W2     = (const float*)d_in[7];
    const float* b2     = (const float*)d_in[8];
    const float* protos = (const float*)d_in[9];

    dim3 grid(8192 / TOK_TILE, 128);   // token tiles x latents
    proto_sae_fused<<<grid, 256, 0, stream>>>(x, W1, b1, gamma, beta, rmean,
                                              rvar, W2, b2, protos,
                                              (float*)d_out);
}
